// EasyTrace_EGNN_70549132804919
// MI455X (gfx1250) — matrix operand order repaired and checked by reading, not compiled, not run on hardware
//
#include <hip/hip_runtime.h>
#include <stdint.h>

typedef __attribute__((ext_vector_type(16))) _Float16 v16h;
typedef __attribute__((ext_vector_type(8)))  _Float16 v8h;
typedef __attribute__((ext_vector_type(8)))  float    v8f;
typedef __attribute__((ext_vector_type(4)))  unsigned int u32x4;
typedef __attribute__((ext_vector_type(8)))  unsigned int u32x8;

namespace {
constexpr int kNodes  = 31;      // nodes per graph (P-1)
constexpr int kEdges  = 930;     // 31*30 directed edges, row-major order
constexpr int kTiles  = 59;      // ceil(930/16)
constexpr int kLayers = 4;

// per-layer weight block layout (bytes) in workspace, mirrored 1:1 in LDS
constexpr int kOffE1T  = 0;      // f16 [64][160] edge_mlp1^T (K 130 -> pad 160)
constexpr int kOffE2T  = 20480;  // f16 [64][64]  edge_mlp2^T
constexpr int kOffC1T  = 28672;  // f16 [64][64]  coord_mlp1^T
constexpr int kOffN1T  = 36864;  // f16 [64][128] node_mlp1^T
constexpr int kOffN2T  = 53248;  // f16 [64][64]  node_mlp2^T
constexpr int kOffC2   = 61440;  // f16 [64]      coord_mlp2 vector
constexpr int kOffBias = 61568;  // f32 e1b,e2b,n1b,n2b,c1b (5*64)
constexpr int kWBytes  = 62848;
constexpr int kWBlock  = 62976;  // 256B-aligned stride between layer blocks
}

struct LayerPtrs {
  const float *e1W, *e1b, *e2W, *e2b, *n1W, *n1b, *n2W, *n2b, *c1W, *c1b, *c2W;
};
struct AllPtrs {
  const float *x, *t, *embW, *embb, *eoW, *eob;
  LayerPtrs L[kLayers];
};

__device__ __forceinline__ float silu_f(float x) { return x / (1.0f + __expf(-x)); }

// A fragment, 16x32 f16: lanes<16 hold M=lane, K=0..15 contiguous;
// lanes>=16 hold K=16..31. rowp points at stage row + chunk*32.
__device__ __forceinline__ v16h frag_a(const _Float16* rowp, int lhalf) {
  v8h a0 = *(const v8h*)(rowp + lhalf * 16);
  v8h a1 = *(const v8h*)(rowp + lhalf * 16 + 8);
  return __builtin_shufflevector(a0, a1, 0, 1, 2, 3, 4, 5, 6, 7,
                                 8, 9, 10, 11, 12, 13, 14, 15);
}

// B fragment, 32x16 f16: lane<16 -> column N=lane, K=c*32+0..15 contiguous;
// lane>=16 -> K=c*32+16..31. Weights stored transposed (column-major in K).
__device__ __forceinline__ v16h frag_b(const _Float16* wt, int kstride, int col,
                                       int c, int lhalf) {
  return *(const v16h*)(wt + (size_t)col * kstride + c * 32 + lhalf * 16);
}

__device__ __forceinline__ v8f wmma16(v16h a, v16h b, v8f c) {
  return __builtin_amdgcn_wmma_f32_16x16x32_f16(false, a, false, b, (short)0, c,
                                                false, false);
}

// ---------------------------------------------------------------------------
// Prep: transpose + convert all layer weights to f16 blocks in workspace.
// ---------------------------------------------------------------------------
__global__ __launch_bounds__(256) void egnn_prep(AllPtrs Pp,
                                                 unsigned char* __restrict__ ws) {
  const int l = blockIdx.x;
  const int tid = threadIdx.x;
  const LayerPtrs Lp = Pp.L[l];
  unsigned char* blk = ws + (size_t)l * kWBlock;
  _Float16* e1t = (_Float16*)(blk + kOffE1T);
  _Float16* e2t = (_Float16*)(blk + kOffE2T);
  _Float16* c1t = (_Float16*)(blk + kOffC1T);
  _Float16* n1t = (_Float16*)(blk + kOffN1T);
  _Float16* n2t = (_Float16*)(blk + kOffN2T);
  _Float16* c2  = (_Float16*)(blk + kOffC2);
  float* bias   = (float*)(blk + kOffBias);

  for (int i = tid; i < 64 * 160; i += 256) {
    int n = i / 160, k = i - n * 160;
    e1t[i] = (k < 130) ? (_Float16)Lp.e1W[k * 64 + n] : (_Float16)0.f;
  }
  for (int i = tid; i < 64 * 64; i += 256) {
    int n = i >> 6, k = i & 63;
    e2t[i] = (_Float16)Lp.e2W[k * 64 + n];
    c1t[i] = (_Float16)Lp.c1W[k * 64 + n];
    n2t[i] = (_Float16)Lp.n2W[k * 64 + n];
  }
  for (int i = tid; i < 64 * 128; i += 256) {
    int n = i >> 7, k = i & 127;
    n1t[i] = (_Float16)Lp.n1W[k * 64 + n];
  }
  if (tid < 64) {
    c2[tid]         = (_Float16)Lp.c2W[tid];
    bias[tid]       = Lp.e1b[tid];
    bias[64 + tid]  = Lp.e2b[tid];
    bias[128 + tid] = Lp.n1b[tid];
    bias[192 + tid] = Lp.n2b[tid];
    bias[256 + tid] = Lp.c1b[tid];
  }
}

// ---------------------------------------------------------------------------
// Main: one workgroup per graph, whole 4-layer EGNN resident in LDS.
// ---------------------------------------------------------------------------
__global__ __launch_bounds__(128) void egnn_main(AllPtrs Pp,
                                                 const unsigned char* __restrict__ ws,
                                                 float* __restrict__ out) {
  __shared__ __align__(32) _Float16 sWB[kWBytes / 2];   // layer weights (TDM dest)
  __shared__ float    sH[32 * 64];                      // node features f32
  __shared__ __align__(32) _Float16 sH16[32 * 64];      // f16 snapshot of h
  __shared__ float    sAgg[32 * 64];                    // message aggregation
  __shared__ float    sEA[kEdges];                      // edge_attr (|d0|^2)
  __shared__ float    sXO[32][4];                       // original positions
  __shared__ float    sC[32][4];                        // current coords
  __shared__ float    sCagg[32][4];                     // coord aggregation
  __shared__ __align__(32) _Float16 sStage[4][16 * 168];// per-wave e_in staging
  __shared__ __align__(32) _Float16 sM[4][16 * 72];     // per-wave m staging
  __shared__ __align__(32) _Float16 sN[32 * 128];       // node-MLP input [h|agg]
  __shared__ __align__(32) _Float16 sN2[2][16 * 72];    // node-MLP mid staging
  __shared__ float    sWn[32];                          // final per-node weights

  const int tid   = threadIdx.x;
  const int w     = tid >> 5;
  const int lane  = tid & 31;
  const int lhalf = lane >> 4;
  const int llo   = lane & 15;
  const int g = blockIdx.x, b = g >> 5, p = g & 31;

  const _Float16* e1t = sWB + (kOffE1T >> 1);
  const _Float16* e2t = sWB + (kOffE2T >> 1);
  const _Float16* c1t = sWB + (kOffC1T >> 1);
  const _Float16* n1t = sWB + (kOffN1T >> 1);
  const _Float16* n2t = sWB + (kOffN2T >> 1);
  const _Float16* c2h = sWB + (kOffC2 >> 1);
  const float* sBias  = (const float*)((const unsigned char*)sWB + kOffBias);

  // ---- init: gather the 31 neighbor positions of slot (b,p)
  if (tid < kNodes) {
    int o = tid + (tid >= p ? 1 : 0);
    const float* xr = Pp.x + (size_t)(b * 32 + o) * 3;
    float x0 = xr[0], x1 = xr[1], x2 = xr[2];
    sXO[tid][0] = x0; sXO[tid][1] = x1; sXO[tid][2] = x2; sXO[tid][3] = 0.f;
    sC[tid][0]  = x0; sC[tid][1]  = x1; sC[tid][2]  = x2; sC[tid][3]  = 0.f;
  }
  __syncthreads();
  for (int e = tid; e < kEdges; e += 128) {
    int r = e / 30, cc = e - r * 30;
    int cn = cc + (cc >= r ? 1 : 0);
    float dx = sXO[r][0] - sXO[cn][0];
    float dy = sXO[r][1] - sXO[cn][1];
    float dz = sXO[r][2] - sXO[cn][2];
    sEA[e] = dx * dx + dy * dy + dz * dz;
  }
  float tval = Pp.t[b];
  if (tid < 64) {
    float hv = tval * Pp.embW[tid] + Pp.embb[tid];
    for (int n = 0; n < 32; ++n) sH[n * 64 + tid] = hv;
  }
  __syncthreads();

  for (int l = 0; l < kLayers; ++l) {
    // ---- TDM: async DMA this layer's prepacked weight block into LDS (wave 0)
    if (w == 0) {
      unsigned lds = (unsigned)(uintptr_t)&sWB[0];
      unsigned long long ga = (unsigned long long)(uintptr_t)(ws + (size_t)l * kWBlock);
      constexpr unsigned ndw = kWBytes / 4;   // DWORD elements
      u32x4 g0;
      g0[0] = 1u;                  // count=1 (valid descriptor), user mode
      g0[1] = lds;                 // lds_addr
      g0[2] = (unsigned)ga;        // global_addr[31:0]
      g0[3] = (unsigned)((ga >> 32) & 0x01FFFFFFull) | (2u << 30); // ga[56:32]|type=2
      u32x8 g1;
      g1[0] = 0x00020000u;                          // data_size = 4B, no multicast
      g1[1] = (ndw & 0xFFFFu) << 16;                // tensor_dim0[15:0]
      g1[2] = ((ndw >> 16) & 0xFFFFu) | (1u << 16); // tensor_dim0[31:16]|dim1=1
      g1[3] = (ndw & 0xFFFFu) << 16;                // tile_dim0
      g1[4] = 1u;                                   // tile_dim1
      g1[5] = ndw;                                  // tensor_dim0_stride[31:0]
      g1[6] = 0u;
      g1[7] = 0u;
      asm volatile("tensor_load_to_lds %0, %1" :: "s"(g0), "s"(g1) : "memory");
    }
    if (w == 1 && l + 1 < kLayers)
      __builtin_prefetch(ws + (size_t)(l + 1) * kWBlock, 0, 1);

    // zero aggregators, snapshot h -> f16 (overlapped with TDM)
    for (int i = tid; i < 2048; i += 128) {
      sAgg[i] = 0.f;
      sH16[i] = (_Float16)sH[i];
    }
    ((float*)sCagg)[tid] = 0.f;
    if (w == 0) __builtin_amdgcn_s_wait_tensorcnt(0);
    __syncthreads();

    // ================= edge tiles (16 edges per tile, 1 tile per wave) ======
    _Float16* stg = sStage[w];
    _Float16* mst = sM[w];
    for (int tile = w; tile < kTiles; tile += 4) {
      const int base = tile * 16;
      const int e = base + llo;
      const bool val = (e < kEdges);
      int r = 0, cn = 0;
      float cdx = 0.f, cdy = 0.f, cdz = 0.f;
      if (val) {
        r = e / 30;
        int cc = e - r * 30;
        cn = cc + (cc >= r ? 1 : 0);
      }
      // stage e_in row: [h[row] | h[col] | radial | edge_attr | 0-pad] (2 lanes/edge)
      if (val) {
        const _Float16* hsrc = sH16 + (lhalf ? cn : r) * 64;
        _Float16* dst = stg + llo * 168 + lhalf * 64;
        #pragma unroll
        for (int j = 0; j < 8; ++j) ((v8h*)dst)[j] = ((const v8h*)hsrc)[j];
        if (!lhalf) {
          float dx = sC[r][0] - sC[cn][0];
          float dy = sC[r][1] - sC[cn][1];
          float dz = sC[r][2] - sC[cn][2];
          float rad = dx * dx + dy * dy + dz * dz;
          float inv = 1.0f / (sqrtf(rad) + 1.0f);
          cdx = dx * inv; cdy = dy * inv; cdz = dz * inv;
          _Float16* rowp = stg + llo * 168;
          rowp[128] = (_Float16)rad;
          rowp[129] = (_Float16)sEA[e];
          unsigned int* zp = (unsigned int*)(rowp + 130);
          #pragma unroll
          for (int j = 0; j < 15; ++j) zp[j] = 0u;   // zero K pad 130..159
        }
      }
      asm volatile("s_wait_dscnt 0x0" ::: "memory");

      // ---- edge_mlp1: [16x160] @ [160x64] + bias, silu
      v8f acc[4];
      #pragma unroll
      for (int t2 = 0; t2 < 4; ++t2) {
        float bv = sBias[t2 * 16 + llo];
        acc[t2] = (v8f){bv, bv, bv, bv, bv, bv, bv, bv};
      }
      #pragma unroll
      for (int c = 0; c < 5; ++c) {
        v16h a = frag_a(stg + llo * 168 + c * 32, lhalf);
        #pragma unroll
        for (int t2 = 0; t2 < 4; ++t2)
          acc[t2] = wmma16(a, frag_b(e1t, 160, t2 * 16 + llo, c, lhalf), acc[t2]);
      }
      #pragma unroll
      for (int t2 = 0; t2 < 4; ++t2) {
        const int col = t2 * 16 + llo;
        #pragma unroll
        for (int v = 0; v < 8; ++v)
          mst[(v + lhalf * 8) * 72 + col] = (_Float16)silu_f(acc[t2][v]);
      }
      asm volatile("s_wait_dscnt 0x0" ::: "memory");

      // ---- edge_mlp2: [16x64] @ [64x64] + bias, m = silu(.)
      v8f acc2[4];
      #pragma unroll
      for (int t2 = 0; t2 < 4; ++t2) {
        float bv = sBias[64 + t2 * 16 + llo];
        acc2[t2] = (v8f){bv, bv, bv, bv, bv, bv, bv, bv};
      }
      #pragma unroll
      for (int c = 0; c < 2; ++c) {
        v16h a = frag_a(mst + llo * 72 + c * 32, lhalf);
        #pragma unroll
        for (int t2 = 0; t2 < 4; ++t2)
          acc2[t2] = wmma16(a, frag_b(e2t, 64, t2 * 16 + llo, c, lhalf), acc2[t2]);
      }
      // silu -> scatter agg (row-major groups of 30) + restage m
      #pragma unroll
      for (int t2 = 0; t2 < 4; ++t2) {
        const int col = t2 * 16 + llo;
        #pragma unroll
        for (int v = 0; v < 8; ++v) {
          const int rowi = v + lhalf * 8;
          float s = silu_f(acc2[t2][v]);
          mst[rowi * 72 + col] = (_Float16)s;
          int ee = base + rowi;
          if (ee < kEdges) {
            int rr = ee / 30;
            atomicAdd(&sAgg[rr * 64 + col], s);
          }
        }
      }
      asm volatile("s_wait_dscnt 0x0" ::: "memory");

      // ---- coord_mlp1: [16x64] @ [64x64] + bias, silu (overwrite staging)
      v8f acc3[4];
      #pragma unroll
      for (int t2 = 0; t2 < 4; ++t2) {
        float bv = sBias[256 + t2 * 16 + llo];
        acc3[t2] = (v8f){bv, bv, bv, bv, bv, bv, bv, bv};
      }
      #pragma unroll
      for (int c = 0; c < 2; ++c) {
        v16h a = frag_a(mst + llo * 72 + c * 32, lhalf);
        #pragma unroll
        for (int t2 = 0; t2 < 4; ++t2)
          acc3[t2] = wmma16(a, frag_b(c1t, 64, t2 * 16 + llo, c, lhalf), acc3[t2]);
      }
      #pragma unroll
      for (int t2 = 0; t2 < 4; ++t2) {
        const int col = t2 * 16 + llo;
        #pragma unroll
        for (int v = 0; v < 8; ++v)
          mst[(v + lhalf * 8) * 72 + col] = (_Float16)silu_f(acc3[t2][v]);
      }
      asm volatile("s_wait_dscnt 0x0" ::: "memory");

      // ---- coord_mlp2 (64->1, no bias) + trans scatter (one lane per edge)
      if (!lhalf && val) {
        const _Float16* mr = mst + llo * 72;
        float s = 0.f;
        #pragma unroll
        for (int j = 0; j < 8; ++j) {
          v8h mv = ((const v8h*)mr)[j];
          v8h cv = ((const v8h*)c2h)[j];
          #pragma unroll
          for (int k = 0; k < 8; ++k) s += (float)mv[k] * (float)cv[k];
        }
        atomicAdd(&sCagg[r][0], cdx * s);
        atomicAdd(&sCagg[r][1], cdy * s);
        atomicAdd(&sCagg[r][2], cdz * s);
      }
    }
    __syncthreads();

    // ================= coord update + node MLP ==============================
    if (tid < 124) {
      int n = tid >> 2, cc = tid & 3;
      if (cc < 3) sC[n][cc] += sCagg[n][cc];
    }
    for (int i = tid; i < 32 * 128; i += 128) {
      int n = i >> 7, k = i & 127;
      _Float16 vv = (_Float16)0.f;
      if (n < kNodes)
        vv = (k < 64) ? sH16[n * 64 + k] : (_Float16)sAgg[n * 64 + (k - 64)];
      sN[i] = vv;
    }
    __syncthreads();

    if (w < 2) {                     // waves 0/1 own node M-tiles 0/1
      const int mt = w;
      v8f acc[4];
      #pragma unroll
      for (int t2 = 0; t2 < 4; ++t2) {
        float bv = sBias[128 + t2 * 16 + llo];
        acc[t2] = (v8f){bv, bv, bv, bv, bv, bv, bv, bv};
      }
      #pragma unroll
      for (int c = 0; c < 4; ++c) {
        v16h a = frag_a(sN + (mt * 16 + llo) * 128 + c * 32, lhalf);
        #pragma unroll
        for (int t2 = 0; t2 < 4; ++t2)
          acc[t2] = wmma16(a, frag_b(n1t, 128, t2 * 16 + llo, c, lhalf), acc[t2]);
      }
      #pragma unroll
      for (int t2 = 0; t2 < 4; ++t2) {
        const int col = t2 * 16 + llo;
        #pragma unroll
        for (int v = 0; v < 8; ++v)
          sN2[mt][(v + lhalf * 8) * 72 + col] = (_Float16)silu_f(acc[t2][v]);
      }
      asm volatile("s_wait_dscnt 0x0" ::: "memory");

      v8f acc2[4];
      #pragma unroll
      for (int t2 = 0; t2 < 4; ++t2) {
        float bv = sBias[192 + t2 * 16 + llo];
        acc2[t2] = (v8f){bv, bv, bv, bv, bv, bv, bv, bv};
      }
      #pragma unroll
      for (int c = 0; c < 2; ++c) {
        v16h a = frag_a(sN2[mt] + llo * 72 + c * 32, lhalf);
        #pragma unroll
        for (int t2 = 0; t2 < 4; ++t2)
          acc2[t2] = wmma16(a, frag_b(n2t, 64, t2 * 16 + llo, c, lhalf), acc2[t2]);
      }
      #pragma unroll
      for (int t2 = 0; t2 < 4; ++t2) {
        const int col = t2 * 16 + llo;
        #pragma unroll
        for (int v = 0; v < 8; ++v) {
          int node = mt * 16 + v + lhalf * 8;
          if (node < kNodes) sH[node * 64 + col] += acc2[t2][v];  // h residual
        }
      }
    }
    __syncthreads();
  }

  // ================= readout =================================================
  if (tid < kNodes) {
    float s = Pp.eob[0];
    #pragma unroll
    for (int k = 0; k < 64; ++k) s += sH[tid * 64 + k] * Pp.eoW[k];
    sWn[tid] = s;
  }
  __syncthreads();
  if (tid < 3) {
    float xc = Pp.x[(size_t)(b * 32 + p) * 3 + tid];
    float acc = 0.f;
    for (int k = 0; k < kNodes; ++k) acc += (xc - sXO[k][tid]) * sWn[k];
    out[(size_t)g * 3 + tid] = acc;
  }
}

// ---------------------------------------------------------------------------
extern "C" void kernel_launch(void* const* d_in, const int* in_sizes, int n_in,
                              void* d_out, int out_size, void* d_ws, size_t ws_size,
                              hipStream_t stream) {
  (void)in_sizes; (void)n_in; (void)out_size; (void)ws_size;
  AllPtrs Pp;
  Pp.x    = (const float*)d_in[0];
  Pp.t    = (const float*)d_in[1];
  Pp.embW = (const float*)d_in[2];
  Pp.embb = (const float*)d_in[3];
  Pp.eoW  = (const float*)d_in[4];
  Pp.eob  = (const float*)d_in[5];
  for (int l = 0; l < kLayers; ++l) {
    const int q = 6 + l * 11;
    Pp.L[l].e1W = (const float*)d_in[q + 0];
    Pp.L[l].e1b = (const float*)d_in[q + 1];
    Pp.L[l].e2W = (const float*)d_in[q + 2];
    Pp.L[l].e2b = (const float*)d_in[q + 3];
    Pp.L[l].n1W = (const float*)d_in[q + 4];
    Pp.L[l].n1b = (const float*)d_in[q + 5];
    Pp.L[l].n2W = (const float*)d_in[q + 6];
    Pp.L[l].n2b = (const float*)d_in[q + 7];
    Pp.L[l].c1W = (const float*)d_in[q + 8];
    Pp.L[l].c1b = (const float*)d_in[q + 9];
    Pp.L[l].c2W = (const float*)d_in[q + 10];
  }
  unsigned char* ws = (unsigned char*)d_ws;
  egnn_prep<<<kLayers, 256, 0, stream>>>(Pp, ws);
  egnn_main<<<32 * 32, 128, 0, stream>>>(Pp, ws, (float*)d_out);
}
